// LinearCoeffGNN_38912403701943
// MI455X (gfx1250) — compile-verified, matrix operand order checked by reading, not run on hardware
//
#include <hip/hip_runtime.h>

typedef __attribute__((ext_vector_type(16))) _Float16 v16h;
typedef __attribute__((ext_vector_type(8)))  _Float16 v8h;
typedef __attribute__((ext_vector_type(8)))  float    v8f;

#define B_   32
#define P_   1024
#define HID_ 512
#define H_   8
#define D_   64
#define MEM_ 64
#define RANK_ 64

// ---------------- Kernel 0: tiny weight contractions ----------------
// A[h][m]  = sum_d w_k[hd]  * w_mem[m,d]
// Gu[r][h] = sum_d w_v[hd]  * w_u[r,hd]   Hu[r][h] = sum_d b_v[hd]*w_u[r,hd]
// Gv[r][h] = sum_d w_v[hd]  * w_v2[r,hd]  Hv[r][h] = sum_d b_v[hd]*w_v2[r,hd]
__global__ void precompute_kernel(const float* __restrict__ w_k,
                                  const float* __restrict__ w_mem,
                                  const float* __restrict__ w_v,
                                  const float* __restrict__ b_v,
                                  const float* __restrict__ w_u,
                                  const float* __restrict__ w_v2,
                                  float* __restrict__ Aw,
                                  float* __restrict__ Gu, float* __restrict__ Hu,
                                  float* __restrict__ Gv, float* __restrict__ Hv) {
  int tid = threadIdx.x;
  for (int i = tid; i < H_ * MEM_; i += blockDim.x) {
    int h = i / MEM_, m = i % MEM_;
    float s = 0.f;
    for (int d = 0; d < D_; ++d) s += w_k[h * D_ + d] * w_mem[m * D_ + d];
    Aw[i] = s;
  }
  for (int i = tid; i < RANK_ * H_; i += blockDim.x) {
    int r = i / H_, h = i % H_;
    float gu = 0.f, hu = 0.f, gv = 0.f, hv = 0.f;
    for (int d = 0; d < D_; ++d) {
      float wu  = w_u[r * HID_ + h * D_ + d];
      float wv2 = w_v2[r * HID_ + h * D_ + d];
      float wv  = w_v[h * D_ + d];
      float bv  = b_v[h * D_ + d];
      gu += wv * wu;  hu += bv * wu;
      gv += wv * wv2; hv += bv * wv2;
    }
    Gu[i] = gu; Hu[i] = hu; Gv[i] = gv; Hv[i] = hv;
  }
}

// ---------------- Kernel 1: softmax statistics per (b,h) ----------------
// S1[b,h,m] = sum_p softmax_p(x[b,p]*A[h,m]) * x[b,p];  T1=sum_m S1; T2=sum_m S1^2
__global__ void softmax_stats_kernel(const float* __restrict__ x,
                                     const float* __restrict__ Aw,
                                     float* __restrict__ S1,
                                     float* __restrict__ T1,
                                     float* __restrict__ T2) {
  __shared__ float xs[P_];
  __shared__ float r0[256];
  __shared__ float r1[256];
  __shared__ float s1sh[MEM_];
  int tid = threadIdx.x;
  int bh  = blockIdx.x;
  int b   = bh >> 3;
  int h   = bh & 7;

  float lmax = -1e30f, lmin = 1e30f;
  for (int i = tid; i < P_; i += 256) {
    float v = x[b * P_ + i];
    xs[i] = v;
    lmax = fmaxf(lmax, v);
    lmin = fminf(lmin, v);
  }
  r0[tid] = lmax; r1[tid] = lmin;
  __syncthreads();
  for (int st = 128; st > 0; st >>= 1) {
    if (tid < st) {
      r0[tid] = fmaxf(r0[tid], r0[tid + st]);
      r1[tid] = fminf(r1[tid], r1[tid + st]);
    }
    __syncthreads();
  }
  float xmax = r0[0], xmin = r1[0];
  __syncthreads();

  int m = tid >> 2, q = tid & 3;
  float Am = Aw[h * MEM_ + m];
  float c  = (Am >= 0.f) ? Am * xmax : Am * xmin;   // max_p of x*A (stable softmax)
  float z = 0.f, s = 0.f;
  int p0 = q * 256;
  for (int i = p0; i < p0 + 256; ++i) {
    float xv = xs[i];
    float e  = __expf(fmaf(xv, Am, -c));
    z += e;
    s = fmaf(xv, e, s);
  }
  r0[tid] = z; r1[tid] = s;
  __syncthreads();
  if (q == 0) {
    float Z = r0[tid] + r0[tid + 1] + r0[tid + 2] + r0[tid + 3];
    float S = r1[tid] + r1[tid + 1] + r1[tid + 2] + r1[tid + 3];
    float s1 = S / Z;
    S1[bh * MEM_ + m] = s1;
    s1sh[m] = s1;
  }
  __syncthreads();
  if (tid == 0) {
    float t1 = 0.f, t2 = 0.f;
    for (int mm = 0; mm < MEM_; ++mm) { float v = s1sh[mm]; t1 += v; t2 += v * v; }
    T1[bh] = t1; T2[bh] = t2;
  }
}

// ---------------- Kernel 2: per-(b,p) alpha/beta -> U, Vp rows (stored f16) ----------------
// One wave32 per (b,p); lanes cover d = lane, lane+32; wave reductions via shfl_xor.
__global__ void uv_kernel(const float* __restrict__ x,
                          const float* __restrict__ w_q, const float* __restrict__ b_q,
                          const float* __restrict__ w_v, const float* __restrict__ b_v,
                          const float* __restrict__ b_u, const float* __restrict__ b_v2,
                          const float* __restrict__ Gu, const float* __restrict__ Hu,
                          const float* __restrict__ Gv, const float* __restrict__ Hv,
                          const float* __restrict__ T1, const float* __restrict__ T2,
                          _Float16* __restrict__ U, _Float16* __restrict__ Vp) {
  int lane = threadIdx.x & 31;
  int wave = threadIdx.x >> 5;
  int idx  = blockIdx.x * 8 + wave;   // flat (b,p)
  int b    = idx >> 10;
  float xv = x[idx];

  float alpha[H_], beta[H_];
#pragma unroll
  for (int h = 0; h < H_; ++h) {
    int i0 = h * D_ + lane, i1 = i0 + 32;
    float u0 = fmaf(xv, w_q[i0], b_q[i0]);
    float u1 = fmaf(xv, w_q[i1], b_q[i1]);
    float q0 = (u0 > 0.f) ? u0 + 1.f : __expf(u0);   // elu(u)+1
    float q1 = (u1 > 0.f) ? u1 + 1.f : __expf(u1);
    float pa = q0 * w_v[i0] + q1 * w_v[i1];
    float pb = q0 * b_v[i0] + q1 * b_v[i1];
#pragma unroll
    for (int msk = 16; msk >= 1; msk >>= 1) {
      pa += __shfl_xor(pa, msk, 32);
      pb += __shfl_xor(pb, msk, 32);
    }
    float t1 = T1[b * H_ + h], t2 = T2[b * H_ + h];
    alpha[h] = pa * t2 + pb * t1;
    beta[h]  = pa * t1 + 64.f * pb;
  }
#pragma unroll
  for (int rr = 0; rr < 2; ++rr) {
    int r = lane + rr * 32;
    float u = b_u[r], v = b_v2[r];
#pragma unroll
    for (int h = 0; h < H_; ++h) {
      u += alpha[h] * Gu[r * H_ + h] + beta[h] * Hu[r * H_ + h];
      v += alpha[h] * Gv[r * H_ + h] + beta[h] * Hv[r * H_ + h];
    }
    U[idx * RANK_ + r]  = (_Float16)u;
    Vp[idx * RANK_ + r] = (_Float16)v;
  }
}

// ---------------- Kernel 3: coeffs = U @ Vp^T per batch, via WMMA f16 ----------------
// 16-bit A/B fragment layout (ISA 7.12.2): lane L, row/col = L%16,
// halves hold K = {klo..klo+7} and {16+klo..16+klo+7}, klo = (L/16)*8.
// Each 8-half run is one 16-byte load (global_load_b128), no per-tile conversion.
__device__ __forceinline__ v16h load_frag_h(const _Float16* __restrict__ row,
                                            int klo, int kbase) {
  const v8h* p0 = reinterpret_cast<const v8h*>(row + kbase + klo);
  const v8h* p1 = reinterpret_cast<const v8h*>(row + kbase + 16 + klo);
  v8h lo = *p0;
  v8h hi = *p1;
  return __builtin_shufflevector(lo, hi, 0, 1, 2, 3, 4, 5, 6, 7,
                                 8, 9, 10, 11, 12, 13, 14, 15);
}

__global__ void coeffs_wmma_kernel(const _Float16* __restrict__ U,
                                   const _Float16* __restrict__ Vp,
                                   float* __restrict__ out) {
  int lane = threadIdx.x & 31;
  int wave = threadIdx.x >> 5;
  int b    = blockIdx.x >> 6;     // batch
  int mt   = blockIdx.x & 63;     // M tile (16 rows)
  int sub  = lane & 15;
  int hi   = lane >> 4;
  int klo  = hi * 8;

  int rowM = mt * 16 + sub;
  const _Float16* arow = U + (size_t)(b * P_ + rowM) * RANK_;
  v16h a0 = load_frag_h(arow, klo, 0);    // K 0..31
  v16h a1 = load_frag_h(arow, klo, 32);   // K 32..63

  float* obase = out + ((size_t)b << 20);    // b * P_*P_
#pragma unroll
  for (int nt = 0; nt < 8; ++nt) {
    int ncol = wave * 128 + nt * 16 + sub;
    const _Float16* brow = Vp + (size_t)(b * P_ + ncol) * RANK_;
    v16h b0 = load_frag_h(brow, klo, 0);
    v16h b1 = load_frag_h(brow, klo, 32);
    v8f c = {};
    c = __builtin_amdgcn_wmma_f32_16x16x32_f16(false, a0, false, b0, (short)0, c, false, false);
    c = __builtin_amdgcn_wmma_f32_16x16x32_f16(false, a1, false, b1, (short)0, c, false, false);
    int rowBase = mt * 16 + hi * 8;
#pragma unroll
    for (int r = 0; r < 8; ++r) {
      obase[(size_t)(rowBase + r) * P_ + ncol] = c[r];
    }
  }
}

extern "C" void kernel_launch(void* const* d_in, const int* in_sizes, int n_in,
                              void* d_out, int out_size, void* d_ws, size_t ws_size,
                              hipStream_t stream) {
  const float* x     = (const float*)d_in[0];
  const float* w_q   = (const float*)d_in[1];
  const float* b_q   = (const float*)d_in[2];
  const float* w_k   = (const float*)d_in[3];
  // d_in[4] (b_k) cancels exactly in the softmax over p — unused.
  const float* w_v   = (const float*)d_in[5];
  const float* b_v   = (const float*)d_in[6];
  const float* w_mem = (const float*)d_in[7];
  const float* w_u   = (const float*)d_in[8];
  const float* b_u   = (const float*)d_in[9];
  const float* w_v2  = (const float*)d_in[10];
  const float* b_v2  = (const float*)d_in[11];
  (void)in_sizes; (void)n_in; (void)out_size; (void)ws_size;

  float* wsf = (float*)d_ws;
  float* Aw = wsf;                       // 512
  float* Gu = Aw + 512;                  // 512
  float* Hu = Gu + 512;
  float* Gv = Hu + 512;
  float* Hv = Gv + 512;
  float* S1 = Hv + 512;                  // B*H*MEM = 16384
  float* T1 = S1 + B_ * H_ * MEM_;       // 256
  float* T2 = T1 + B_ * H_;              // 256
  // f16 U/Vp after the f32 stats region (16-byte aligned: 19456 floats in).
  _Float16* Uh = (_Float16*)(T2 + B_ * H_);         // B*P*RANK halves (4 MB)
  _Float16* Vh = Uh + (size_t)B_ * P_ * RANK_;      // 4 MB

  precompute_kernel<<<1, 256, 0, stream>>>(w_k, w_mem, w_v, b_v, w_u, w_v2,
                                           Aw, Gu, Hu, Gv, Hv);
  softmax_stats_kernel<<<B_ * H_, 256, 0, stream>>>(x, Aw, S1, T1, T2);
  uv_kernel<<<(B_ * P_) / 8, 256, 0, stream>>>(x, w_q, b_q, w_v, b_v, b_u, b_v2,
                                               Gu, Hu, Gv, Hv, T1, T2, Uh, Vh);
  coeffs_wmma_kernel<<<B_ * 64, 256, 0, stream>>>(Uh, Vh, (float*)d_out);
}
